// CliffordTP_84164179132822
// MI455X (gfx1250) — matrix operand order repaired and checked by reading, not compiled, not run on hardware
//
#include <hip/hip_runtime.h>

// ---------------------------------------------------------------------------
// Clifford geometric-product MLP (Cl(3,0), 8 blades) as 3 chained GEMMs:
//   act_{l+1}[2048,4096] = act_l[2048,4096] @ Wbig_l[4096,4096]
//   Wbig[(i,p),(o,k)] = sign(p, p^k) * w[o,i,p^k]
// fp32-accurate 3-term bf16 split on V_WMMA_F32_16X16X32_BF16.
// Conversion/expansion hoisted out of the GEMM; GEMM staging uses the CDNA5
// async global->LDS path (ASYNCcnt) so the inner loop is WMMA-bound.
// ---------------------------------------------------------------------------

typedef __attribute__((ext_vector_type(16))) __bf16        v16bf;
typedef __attribute__((ext_vector_type(8)))  float         v8f;
typedef __attribute__((ext_vector_type(4)))  float         f32x4;
typedef __attribute__((ext_vector_type(4)))  unsigned int  u32x4;
typedef __attribute__((ext_vector_type(2)))  unsigned int  u32x2;

#define KDIM 4096            // K = I*8 = N = O*8
#define MDIM 2048            // batch rows
#define PLANE (MDIM * KDIM)          // 8,388,608 f32 activation plane
#define BELE  ((size_t)KDIM * KDIM)  // 16,777,216 expanded weight elements

// bf16 round-to-nearest-even high bits of an f32
__device__ __forceinline__ unsigned int bf16_hi_bits(float v) {
  unsigned int b = __float_as_uint(v);
  return (b + 0x7FFFu + ((b >> 16) & 1u)) >> 16;
}
__device__ __forceinline__ void split2(float v, unsigned int& h, unsigned int& l) {
  h = bf16_hi_bits(v);
  l = bf16_hi_bits(v - __uint_as_float(h << 16));
}

// CDNA5 async global->LDS copy (16B per lane), tracked by ASYNCcnt.
__device__ __forceinline__ void async_copy_b128(unsigned lds_off, const void* gaddr) {
  asm volatile("global_load_async_to_lds_b128 %0, %1, off"
               :: "v"(lds_off), "v"(gaddr) : "memory");
}
__device__ __forceinline__ void wait_asynccnt0() {
  asm volatile("s_wait_asynccnt 0" ::: "memory");
}

// ---------------------------------------------------------------------------
// Preprocess 1: f32 plane -> bf16 hi/lo planes (4 f32/thread)
// ---------------------------------------------------------------------------
__global__ __launch_bounds__(256)
void split_f32_bf16(const float* __restrict__ src,
                    unsigned short* __restrict__ hi,
                    unsigned short* __restrict__ lo) {
  const size_t idx = (size_t)blockIdx.x * 256 + threadIdx.x; // f32x4 index
  f32x4 v = ((const f32x4*)src)[idx];
  unsigned int h[4], l[4];
#pragma unroll
  for (int j = 0; j < 4; ++j) split2(v[j], h[j], l[j]);
  u32x2 hp, lp;
  hp[0] = h[0] | (h[1] << 16); hp[1] = h[2] | (h[3] << 16);
  lp[0] = l[0] | (l[1] << 16); lp[1] = l[2] | (l[3] << 16);
  ((u32x2*)hi)[idx] = hp;
  ((u32x2*)lo)[idx] = lp;
}

// ---------------------------------------------------------------------------
// Preprocess 2: expand Cayley-structured Wbig, N-major [n=(o,kb)][k=(i,p)],
// as bf16 hi/lo.  Wbig[n][k] = sign(p, p^kb) * w[o, i, p^kb]
// ---------------------------------------------------------------------------
__global__ __launch_bounds__(256)
void expand_cayley(const float* __restrict__ W,        // [512, 512, 8]
                   unsigned short* __restrict__ bhi,   // [4096, 4096]
                   unsigned short* __restrict__ blo) {
  const size_t gid = (size_t)blockIdx.x * 256 + threadIdx.x;
  const int n  = (int)(gid >> 9);     // output column 0..4095
  const int ig = (int)(gid & 511);    // input-channel group 0..511
  const int o  = n >> 3;
  const int kb = n & 7;
  const float* wg = W + ((size_t)o * 512 + ig) * 8;
  unsigned int hb[8], lb[8];
#pragma unroll
  for (int p = 0; p < 8; ++p) {
    const int q = p ^ kb;
    float v = wg[q];
    const int neg = (__popc((p >> 1) & q) + __popc((p >> 2) & q)) & 1;
    v = neg ? -v : v;
    split2(v, hb[p], lb[p]);
  }
  u32x4 hv, lv;
#pragma unroll
  for (int j = 0; j < 4; ++j) {
    hv[j] = hb[2 * j] | (hb[2 * j + 1] << 16);
    lv[j] = lb[2 * j] | (lb[2 * j + 1] << 16);
  }
  const size_t off = (size_t)n * KDIM + (size_t)ig * 8;
  *(u32x4*)&bhi[off] = hv;
  *(u32x4*)&blo[off] = lv;
}

// ---------------------------------------------------------------------------
// Main GEMM on preprocessed bf16 hi/lo operands.
// 128x128 tile / block, 8 waves (2x4), 4x2 16x16 fragments per wave.
// Double-buffered LDS (80 KB dynamic), async global->LDS staging,
// one barrier per K-step.
// ---------------------------------------------------------------------------
#define LDS_ROW 40                    // padded bf16 row stride (80 B)
#define REGION  (128 * LDS_ROW)       // 5120 ushorts per matrix region
#define BUFSZ   (4 * REGION)          // Ah | Al | Bh | Bl

__global__ __launch_bounds__(256)
void clifford_gemm_pre(const unsigned short* __restrict__ Ah,
                       const unsigned short* __restrict__ Al,
                       const unsigned short* __restrict__ Bh,
                       const unsigned short* __restrict__ Bl,
                       float* __restrict__ D) {
  extern __shared__ unsigned short smem[];   // 2 * BUFSZ ushorts = 80 KB

  const int t     = threadIdx.x;
  const int lane  = t & 31;
  const int wid   = t >> 5;
  const int waveM = wid >> 2;          // 0..1
  const int waveN = wid & 3;           // 0..3
  const int n0    = blockIdx.x * 128;
  const int m0    = blockIdx.y * 128;

  // staging: each thread owns half a tile row (16 bf16 = 32 B)
  const int rowS = t >> 1;
  const int c0   = (t & 1) * 16;
  const unsigned short* gAh = Ah + (size_t)(m0 + rowS) * KDIM + c0;
  const unsigned short* gAl = Al + (size_t)(m0 + rowS) * KDIM + c0;
  const unsigned short* gBh = Bh + (size_t)(n0 + rowS) * KDIM + c0;
  const unsigned short* gBl = Bl + (size_t)(n0 + rowS) * KDIM + c0;
  const int sOff = rowS * LDS_ROW + c0;

  v8f acc[4][2];
  {
    v8f z = {0.f, 0.f, 0.f, 0.f, 0.f, 0.f, 0.f, 0.f};
#pragma unroll
    for (int mf = 0; mf < 4; ++mf)
#pragma unroll
      for (int nf = 0; nf < 2; ++nf) acc[mf][nf] = z;
  }

  // ---- async-stage tile k0 into buffer buf (no VGPR round-trip) ----
  auto stage = [&](int buf, int k0) {
    const unsigned short* srcs[4] = {gAh + k0, gAl + k0, gBh + k0, gBl + k0};
    // wave-relative LDS byte offset == low 32 bits of the generic pointer
    const unsigned base =
        (unsigned)(size_t)(const void*)(smem + (size_t)buf * BUFSZ + sOff);
#pragma unroll
    for (int r = 0; r < 4; ++r) {
      async_copy_b128(base + r * (REGION * 2), srcs[r]);
      async_copy_b128(base + r * (REGION * 2) + 16, (const char*)srcs[r] + 16);
    }
  };

  stage(0, 0);
  __builtin_prefetch(gAh + 32, 0, 1);
  __builtin_prefetch(gBh + 32, 0, 1);

  int buf = 0;
  for (int k0 = 0; k0 < KDIM; k0 += 32, buf ^= 1) {
    wait_asynccnt0();                      // my async writes into buf done
    __syncthreads();                       // buffer published to all waves
    if (k0 + 32 < KDIM) {
      stage(buf ^ 1, k0 + 32);             // overlap next-tile staging
      __builtin_prefetch(gAh + k0 + 64, 0, 1);
      __builtin_prefetch(gBh + k0 + 64, 0, 1);
    }

    // ---- fragments + WMMA from buffer buf ----
    const unsigned short* bAh = smem + buf * BUFSZ;
    const unsigned short* bAl = bAh + REGION;
    const unsigned short* bBh = bAl + REGION;
    const unsigned short* bBl = bBh + REGION;
    const int half = lane >> 4;
    const int l16  = lane & 15;
    union Fr { v16bf v; u32x4 q[2]; };

    Fr ah[4], al[4];
#pragma unroll
    for (int mf = 0; mf < 4; ++mf) {
      const int r = (waveM * 64 + mf * 16 + l16) * LDS_ROW;
      // A 16x32: elems 0..7 = K[8*half..), elems 8..15 = K[16+8*half..)
      ah[mf].q[0] = *(const u32x4*)&bAh[r + half * 8];
      ah[mf].q[1] = *(const u32x4*)&bAh[r + 16 + half * 8];
      al[mf].q[0] = *(const u32x4*)&bAl[r + half * 8];
      al[mf].q[1] = *(const u32x4*)&bAl[r + 16 + half * 8];
    }
#pragma unroll
    for (int nf = 0; nf < 2; ++nf) {
      const int r = (waveN * 32 + nf * 16 + l16) * LDS_ROW + half * 8;
      Fr bh, bl;
      bh.q[0] = *(const u32x4*)&bBh[r];
      bh.q[1] = *(const u32x4*)&bBh[r + 16];
      bl.q[0] = *(const u32x4*)&bBl[r];
      bl.q[1] = *(const u32x4*)&bBl[r + 16];
#pragma unroll
      for (int mf = 0; mf < 4; ++mf) {
        acc[mf][nf] = __builtin_amdgcn_wmma_f32_16x16x32_bf16(
            false, ah[mf].v, false, bh.v, (short)0, acc[mf][nf], false, false);
        acc[mf][nf] = __builtin_amdgcn_wmma_f32_16x16x32_bf16(
            false, ah[mf].v, false, bl.v, (short)0, acc[mf][nf], false, false);
        acc[mf][nf] = __builtin_amdgcn_wmma_f32_16x16x32_bf16(
            false, al[mf].v, false, bh.v, (short)0, acc[mf][nf], false, false);
      }
    }
  }

  // ---- store: VGPR r -> M = r + 8*(lane>=16), N = lane%16 ----
  {
    const int half = lane >> 4;
    const int l16  = lane & 15;
#pragma unroll
    for (int mf = 0; mf < 4; ++mf)
#pragma unroll
      for (int nf = 0; nf < 2; ++nf) {
        const int row0 = m0 + waveM * 64 + mf * 16 + half * 8;
        const int col  = n0 + waveN * 32 + nf * 16 + l16;
        float* dp = D + (size_t)row0 * KDIM + col;
        v8f c = acc[mf][nf];
#pragma unroll
        for (int r = 0; r < 8; ++r) dp[(size_t)r * KDIM] = c[r];
      }
  }
}

// ---------------------------------------------------------------------------
// Fallback: fully fused kernel (no workspace).
// ---------------------------------------------------------------------------
__global__ __launch_bounds__(256)
void clifford_gemm_fused(const float* __restrict__ A, const float* __restrict__ W,
                         float* __restrict__ D) {
  __shared__ unsigned short sAh[REGION], sAl[REGION], sBh[REGION], sBl[REGION];
  const int t = threadIdx.x, lane = t & 31, wid = t >> 5;
  const int waveM = wid >> 2, waveN = wid & 3;
  const int n0 = blockIdx.x * 128, m0 = blockIdx.y * 128;
  v8f acc[4][2];
  {
    v8f z = {0.f, 0.f, 0.f, 0.f, 0.f, 0.f, 0.f, 0.f};
#pragma unroll
    for (int mf = 0; mf < 4; ++mf)
#pragma unroll
      for (int nf = 0; nf < 2; ++nf) acc[mf][nf] = z;
  }
  const int rowS = t >> 1, c0 = (t & 1) * 16;
  const int gn = n0 + rowS, o = gn >> 3, kb = gn & 7;
  const float* wrow = W + (size_t)o * KDIM;
  const float* arow = A + (size_t)(m0 + rowS) * KDIM;

  for (int k0 = 0; k0 < KDIM; k0 += 32) {
    {
      const f32x4* ap = (const f32x4*)(arow + k0 + c0);
      f32x4 av[4];
#pragma unroll
      for (int j = 0; j < 4; ++j) av[j] = ap[j];
      unsigned int hb[16], lb[16];
#pragma unroll
      for (int j = 0; j < 16; ++j) split2(av[j >> 2][j & 3], hb[j], lb[j]);
      u32x4 hv[2], lv[2];
#pragma unroll
      for (int j = 0; j < 8; ++j) {
        hv[j >> 2][j & 3] = hb[2 * j] | (hb[2 * j + 1] << 16);
        lv[j >> 2][j & 3] = lb[2 * j] | (lb[2 * j + 1] << 16);
      }
      *(u32x4*)&sAh[rowS * LDS_ROW + c0] = hv[0];
      *(u32x4*)&sAh[rowS * LDS_ROW + c0 + 8] = hv[1];
      *(u32x4*)&sAl[rowS * LDS_ROW + c0] = lv[0];
      *(u32x4*)&sAl[rowS * LDS_ROW + c0 + 8] = lv[1];
    }
#pragma unroll
    for (int g = 0; g < 2; ++g) {
      const float* wg = wrow + k0 + c0 + g * 8;
      unsigned int hb[8], lb[8];
#pragma unroll
      for (int p = 0; p < 8; ++p) {
        const int q = p ^ kb;
        float v = wg[q];
        const int neg = (__popc((p >> 1) & q) + __popc((p >> 2) & q)) & 1;
        split2(neg ? -v : v, hb[p], lb[p]);
      }
      u32x4 hv, lv;
#pragma unroll
      for (int j = 0; j < 4; ++j) {
        hv[j] = hb[2 * j] | (hb[2 * j + 1] << 16);
        lv[j] = lb[2 * j] | (lb[2 * j + 1] << 16);
      }
      *(u32x4*)&sBh[rowS * LDS_ROW + c0 + g * 8] = hv;
      *(u32x4*)&sBl[rowS * LDS_ROW + c0 + g * 8] = lv;
    }
    __syncthreads();
    const int half = lane >> 4, l16 = lane & 15;
    union Fr { v16bf v; u32x4 q[2]; };
    Fr ah[4], al[4];
#pragma unroll
    for (int mf = 0; mf < 4; ++mf) {
      const int r = (waveM * 64 + mf * 16 + l16) * LDS_ROW;
      ah[mf].q[0] = *(const u32x4*)&sAh[r + half * 8];
      ah[mf].q[1] = *(const u32x4*)&sAh[r + 16 + half * 8];
      al[mf].q[0] = *(const u32x4*)&sAl[r + half * 8];
      al[mf].q[1] = *(const u32x4*)&sAl[r + 16 + half * 8];
    }
#pragma unroll
    for (int nf = 0; nf < 2; ++nf) {
      const int r = (waveN * 32 + nf * 16 + l16) * LDS_ROW + half * 8;
      Fr bh, bl;
      bh.q[0] = *(const u32x4*)&sBh[r];
      bh.q[1] = *(const u32x4*)&sBh[r + 16];
      bl.q[0] = *(const u32x4*)&sBl[r];
      bl.q[1] = *(const u32x4*)&sBl[r + 16];
#pragma unroll
      for (int mf = 0; mf < 4; ++mf) {
        acc[mf][nf] = __builtin_amdgcn_wmma_f32_16x16x32_bf16(
            false, ah[mf].v, false, bh.v, (short)0, acc[mf][nf], false, false);
        acc[mf][nf] = __builtin_amdgcn_wmma_f32_16x16x32_bf16(
            false, ah[mf].v, false, bl.v, (short)0, acc[mf][nf], false, false);
        acc[mf][nf] = __builtin_amdgcn_wmma_f32_16x16x32_bf16(
            false, al[mf].v, false, bh.v, (short)0, acc[mf][nf], false, false);
      }
    }
    __syncthreads();
  }
  const int half = lane >> 4, l16 = lane & 15;
#pragma unroll
  for (int mf = 0; mf < 4; ++mf)
#pragma unroll
    for (int nf = 0; nf < 2; ++nf) {
      const int row0 = m0 + waveM * 64 + mf * 16 + half * 8;
      const int col = n0 + waveN * 32 + nf * 16 + l16;
      float* dp = D + (size_t)row0 * KDIM + col;
      v8f c = acc[mf][nf];
#pragma unroll
      for (int r = 0; r < 8; ++r) dp[(size_t)r * KDIM] = c[r];
    }
}

// ---------------------------------------------------------------------------
extern "C" void kernel_launch(void* const* d_in, const int* in_sizes, int n_in,
                              void* d_out, int out_size, void* d_ws, size_t ws_size,
                              hipStream_t stream) {
  (void)in_sizes; (void)n_in; (void)out_size;
  const float* x = (const float*)d_in[0];
  const float* w[3] = {(const float*)d_in[1], (const float*)d_in[2], (const float*)d_in[3]};
  float* out = (float*)d_out;
  const size_t plane = (size_t)PLANE;

  hipMemcpyAsync(out, x, plane * sizeof(float), hipMemcpyDeviceToDevice, stream);

  dim3 grid(KDIM / 128, MDIM / 128), block(256);
  const size_t needed = (2 * (size_t)PLANE + 2 * BELE) * sizeof(unsigned short); // 96 MB

  if (ws_size >= needed) {
    unsigned short* Ahi = (unsigned short*)d_ws;
    unsigned short* Alo = Ahi + PLANE;
    unsigned short* Bhi = Alo + PLANE;
    unsigned short* Blo = Bhi + BELE;
    const int preGrid = PLANE / (256 * 4);
    const int expGrid = (int)(BELE / (256 * 8));

    split_f32_bf16<<<preGrid, 256, 0, stream>>>(x, Ahi, Alo);
    for (int l = 0; l < 3; ++l) {
      expand_cayley<<<expGrid, 256, 0, stream>>>(w[l], Bhi, Blo);
      clifford_gemm_pre<<<grid, block, 2 * BUFSZ * sizeof(unsigned short), stream>>>(
          Ahi, Alo, Bhi, Blo, out + (l + 1) * plane);
      if (l < 2)
        split_f32_bf16<<<preGrid, 256, 0, stream>>>(out + (l + 1) * plane, Ahi, Alo);
    }
  } else {
    clifford_gemm_fused<<<grid, block, 0, stream>>>(out, w[0], out + plane);
    clifford_gemm_fused<<<grid, block, 0, stream>>>(out + plane, w[1], out + 2 * plane);
    clifford_gemm_fused<<<grid, block, 0, stream>>>(out + 2 * plane, w[2], out + 3 * plane);
  }
}
